// EdgeMLP_76390288327309
// MI455X (gfx1250) — matrix operand order repaired and checked by reading, not compiled, not run on hardware
//
#include <hip/hip_runtime.h>

// ---------------------------------------------------------------------------
// EdgeMLP fused kernel for gfx1250 (MI455X, wave32, WMMA)
//   cat = [efeat | src_feat[src_idx] | dst_feat[dst_idx]]   [E,384] (f16 in LDS)
//   h1  = SiLU(cat @ W1 + b1)   [E,512]   (f16 in LDS)
//   h2  = SiLU(h1  @ W2 + b2)   [E,512]   (f16 in LDS)
//   y   = h2 @ W3 + b3          [E,128]   (f32 in LDS)
//   out = LayerNorm(y) * gamma + beta     (f32 to HBM)
// GEMMs run on v_wmma_f32_16x16x32_f16, f32 accumulate.
// ---------------------------------------------------------------------------

typedef __attribute__((ext_vector_type(16))) _Float16 v16h;
typedef __attribute__((ext_vector_type(8)))  _Float16 v8h;
typedef __attribute__((ext_vector_type(4)))  _Float16 v4h;
typedef __attribute__((ext_vector_type(8)))  float    v8f;
typedef __attribute__((ext_vector_type(4)))  float    v4f;

#define DD    128
#define HH    512
#define CATD  384
#define BM    128          // edges per workgroup (8 row-tiles of 16)
#define NTHR  512          // 16 waves (wave32)
#define SA    392          // cat LDS row stride (halves), padded (+8) vs 384
#define SH    520          // hidden LDS row stride (halves), padded (+8) vs 512
#define YS    132          // y LDS row stride (floats), padded (+4) vs 128

// LDS layout (bytes). bufA: cat(100,352B) then h2(133,120B). bufB: h1(133,120B) then y(67,584B)+stats.
#define OFF_A   0
#define OFF_B   133120
#define OFF_PS  (OFF_B + 67584)     // psum[4][128] f32
#define OFF_PQ  (OFF_PS + 2048)     // psq [4][128] f32
#define OFF_MU  (OFF_PQ + 2048)     // mu  [128]    f32
#define OFF_RS  (OFF_MU + 512)      // rstd[128]    f32
#define LDS_BYTES 266240            // <= 320KB per WGP

// --------------------------- weight pre-swizzle ----------------------------
// Pack W[K][N] (f32, row-major) into f16 B-fragments for V_WMMA_*_16x16x32:
//   fragment (kt,ct) = 32 lanes x 16 halves; lane l covers column ct*16+(l&15),
//   element t covers K = kt*32 + 16*(t>>3) + 8*(l>>4) + (t&7)   (ISA 7.12.2, 16-bit B 32x16).
// Flat layout: out[((kt*CT+ct)*32 + l)*16 + t]  -> per-lane 32B coalesced loads.
__global__ void swz_kernel(const float* __restrict__ W, _Float16* __restrict__ out,
                           int K, int N, int CT) {
    int tid = blockIdx.x * blockDim.x + threadIdx.x;
    if (tid >= K * N) return;
    int t    = tid & 15;
    int l    = (tid >> 4) & 31;
    int frag = tid >> 9;
    int ct   = frag % CT;
    int kt   = frag / CT;
    int n    = ct * 16 + (l & 15);
    int k    = kt * 32 + ((t >> 3) << 4) + ((l >> 4) << 3) + (t & 7);
    out[tid] = (_Float16)W[(size_t)k * N + n];
}

// ------------------------------- helpers -----------------------------------
__device__ __forceinline__ v8f wmma_f16(v16h a, v16h b, v8f c) {
    return __builtin_amdgcn_wmma_f32_16x16x32_f16(false, a, false, b, (short)0, c,
                                                  false, false);
}

// A fragment (16x32, f16) from row-major LDS. Lane l: row = rowBase + (l&15);
// elements 0..7 -> K = kBase + 8*(l>>4) + 0..7 ; elements 8..15 -> +16.
// Two 16B ds_load_b128 per lane; padded strides keep these conflict-free.
__device__ __forceinline__ v16h load_a(const _Float16* base, int strideH,
                                       int rowBase, int kBase, int lane) {
    const _Float16* p = base + (size_t)(rowBase + (lane & 15)) * strideH
                             + kBase + ((lane >> 4) << 3);
    v8h lo = *(const v8h*)p;
    v8h hi = *(const v8h*)(p + 16);
    v16h r;
#pragma unroll
    for (int i = 0; i < 8; ++i) { r[i] = lo[i]; r[i + 8] = hi[i]; }
    return r;
}

// B fragment from pre-swizzled weights: 1024B/fragment, 32B per lane.
__device__ __forceinline__ v16h load_b(const _Float16* wswz, int frag, int lane) {
    return *(const v16h*)(wswz + ((size_t)frag << 9) + (lane << 4));
}

__device__ __forceinline__ float silu(float x) {
    return x * (1.0f / (1.0f + __expf(-x)));
}

// ------------------------------ main kernel --------------------------------
__global__ __launch_bounds__(NTHR, 1)
void edge_mlp_kernel(const float* __restrict__ efeat,
                     const float* __restrict__ srcf,
                     const float* __restrict__ dstf,
                     const int*   __restrict__ sidx,
                     const int*   __restrict__ didx,
                     const float* __restrict__ b1,
                     const float* __restrict__ b2,
                     const float* __restrict__ b3,
                     const float* __restrict__ gamma,
                     const float* __restrict__ beta,
                     const _Float16* __restrict__ W1s,
                     const _Float16* __restrict__ W2s,
                     const _Float16* __restrict__ W3s,
                     float* __restrict__ out) {
    extern __shared__ char smem[];
    _Float16* bufA = (_Float16*)(smem + OFF_A);   // cat, later h2
    _Float16* h1   = (_Float16*)(smem + OFF_B);   // h1
    float*    ybuf = (float*)   (smem + OFF_B);   // y (after h1 is dead)
    float*    psum = (float*)   (smem + OFF_PS);
    float*    psq  = (float*)   (smem + OFF_PQ);
    float*    mu   = (float*)   (smem + OFF_MU);
    float*    rs   = (float*)   (smem + OFF_RS);

    const int  tid   = threadIdx.x;
    const int  lane  = tid & 31;
    const int  wave  = tid >> 5;
    const long eBase = (long)blockIdx.x * BM;
    const int  g     = lane >> 4;
    const int  nl    = lane & 15;

    // ---- stage 0: gather + concat + f32->f16, into LDS (coalesced float4) ----
#pragma unroll
    for (int s = 0; s < 3; ++s) {
        const float* sp = (s == 0) ? efeat : ((s == 1) ? srcf : dstf);
#pragma unroll
        for (int i = 0; i < 8; ++i) {            // 128*128 floats / (512 thr * 4)
            int  flat = (i * NTHR + tid) * 4;
            int  r    = flat >> 7;
            int  c    = flat & 127;
            long row  = (s == 0) ? (eBase + r)
                                 : (long)((s == 1) ? sidx[eBase + r] : didx[eBase + r]);
            v4f v = *(const v4f*)(sp + row * DD + c);
            v4h h;
            h[0] = (_Float16)v[0]; h[1] = (_Float16)v[1];
            h[2] = (_Float16)v[2]; h[3] = (_Float16)v[3];
            *(v4h*)(bufA + (size_t)r * SA + s * DD + c) = h;
        }
    }
    __syncthreads();

    // ---- layer 1: h1 = SiLU(cat @ W1 + b1), K=384 (12 kt), N=512 ----
    // wave owns 2 column tiles; accumulators for all 8 row tiles stay live so
    // each B fragment is loaded exactly once per wave per kt.
    {
        const int ct0 = wave * 2, ct1 = ct0 + 1;
        v8f acc[8][2];
#pragma unroll
        for (int rt = 0; rt < 8; ++rt) {
            acc[rt][0] = (v8f){0.f,0.f,0.f,0.f,0.f,0.f,0.f,0.f};
            acc[rt][1] = (v8f){0.f,0.f,0.f,0.f,0.f,0.f,0.f,0.f};
        }
        for (int kt = 0; kt < CATD / 32; ++kt) {
            v16h bf0 = load_b(W1s, kt * 32 + ct0, lane);
            v16h bf1 = load_b(W1s, kt * 32 + ct1, lane);
#pragma unroll
            for (int rt = 0; rt < 8; ++rt) {
                v16h a = load_a(bufA, SA, rt * 16, kt * 32, lane);
                acc[rt][0] = wmma_f16(a, bf0, acc[rt][0]);
                acc[rt][1] = wmma_f16(a, bf1, acc[rt][1]);
            }
        }
        float bias0 = b1[ct0 * 16 + nl];
        float bias1 = b1[ct1 * 16 + nl];
#pragma unroll
        for (int rt = 0; rt < 8; ++rt) {
#pragma unroll
            for (int v = 0; v < 8; ++v) {        // C layout: row = v + 8*g
                int row = rt * 16 + v + 8 * g;
                h1[(size_t)row * SH + ct0 * 16 + nl] = (_Float16)silu(acc[rt][0][v] + bias0);
                h1[(size_t)row * SH + ct1 * 16 + nl] = (_Float16)silu(acc[rt][1][v] + bias1);
            }
        }
    }
    __syncthreads();

    // ---- layer 2: h2 = SiLU(h1 @ W2 + b2), K=512 (16 kt), N=512; h2 -> bufA ----
    {
        const int ct0 = wave * 2, ct1 = ct0 + 1;
        v8f acc[8][2];
#pragma unroll
        for (int rt = 0; rt < 8; ++rt) {
            acc[rt][0] = (v8f){0.f,0.f,0.f,0.f,0.f,0.f,0.f,0.f};
            acc[rt][1] = (v8f){0.f,0.f,0.f,0.f,0.f,0.f,0.f,0.f};
        }
        for (int kt = 0; kt < HH / 32; ++kt) {
            v16h bf0 = load_b(W2s, kt * 32 + ct0, lane);
            v16h bf1 = load_b(W2s, kt * 32 + ct1, lane);
#pragma unroll
            for (int rt = 0; rt < 8; ++rt) {
                v16h a = load_a(h1, SH, rt * 16, kt * 32, lane);
                acc[rt][0] = wmma_f16(a, bf0, acc[rt][0]);
                acc[rt][1] = wmma_f16(a, bf1, acc[rt][1]);
            }
        }
        float bias0 = b2[ct0 * 16 + nl];
        float bias1 = b2[ct1 * 16 + nl];
#pragma unroll
        for (int rt = 0; rt < 8; ++rt) {
#pragma unroll
            for (int v = 0; v < 8; ++v) {
                int row = rt * 16 + v + 8 * g;
                bufA[(size_t)row * SH + ct0 * 16 + nl] = (_Float16)silu(acc[rt][0][v] + bias0);
                bufA[(size_t)row * SH + ct1 * 16 + nl] = (_Float16)silu(acc[rt][1][v] + bias1);
            }
        }
    }
    __syncthreads();

    // ---- layer 3: y = h2 @ W3 + b3, K=512 (16 kt), N=128; y (f32) -> ybuf ----
    {
        const int ct  = wave & 7;            // 8 column tiles
        const int rts = (wave >> 3) * 4;     // two row-halves of 4 row tiles
        v8f acc[4];
#pragma unroll
        for (int r4 = 0; r4 < 4; ++r4)
            acc[r4] = (v8f){0.f,0.f,0.f,0.f,0.f,0.f,0.f,0.f};
        for (int kt = 0; kt < HH / 32; ++kt) {
            v16h bf = load_b(W3s, kt * 8 + ct, lane);
#pragma unroll
            for (int r4 = 0; r4 < 4; ++r4) {
                v16h a = load_a(bufA, SH, (rts + r4) * 16, kt * 32, lane);
                acc[r4] = wmma_f16(a, bf, acc[r4]);
            }
        }
        float bias = b3[ct * 16 + nl];
#pragma unroll
        for (int r4 = 0; r4 < 4; ++r4) {
#pragma unroll
            for (int v = 0; v < 8; ++v) {
                int row = (rts + r4) * 16 + v + 8 * g;
                ybuf[(size_t)row * YS + ct * 16 + nl] = acc[r4][v] + bias;
            }
        }
    }
    __syncthreads();

    // ---- LayerNorm over D=128 per row, then coalesced store ----
    {
        int r = tid & 127, seg = tid >> 7;   // 4 segments of 32 cols per row
        const float* yr = ybuf + (size_t)r * YS + seg * 32;
        float s = 0.f, q = 0.f;
#pragma unroll
        for (int c = 0; c < 32; c += 4) {
            v4f v = *(const v4f*)(yr + c);
            s += v[0] + v[1] + v[2] + v[3];
            q += v[0] * v[0] + v[1] * v[1] + v[2] * v[2] + v[3] * v[3];
        }
        psum[seg * 128 + r] = s;
        psq [seg * 128 + r] = q;
    }
    __syncthreads();
    if (tid < 128) {
        float s = psum[tid] + psum[128 + tid] + psum[256 + tid] + psum[384 + tid];
        float q = psq [tid] + psq [128 + tid] + psq [256 + tid] + psq [384 + tid];
        float m   = s * (1.0f / 128.0f);
        float var = q * (1.0f / 128.0f) - m * m;
        mu[tid] = m;
        rs[tid] = rsqrtf(var + 1e-5f);
    }
    __syncthreads();
#pragma unroll
    for (int i = 0; i < 8; ++i) {
        int flat = (i * NTHR + tid) * 4;
        int r = flat >> 7, c = flat & 127;
        v4f v  = *(const v4f*)(ybuf + (size_t)r * YS + c);
        v4f gv = *(const v4f*)(gamma + c);
        v4f bv = *(const v4f*)(beta + c);
        float m = mu[r], ir = rs[r];
        v4f o;
        o[0] = (v[0] - m) * ir * gv[0] + bv[0];
        o[1] = (v[1] - m) * ir * gv[1] + bv[1];
        o[2] = (v[2] - m) * ir * gv[2] + bv[2];
        o[3] = (v[3] - m) * ir * gv[3] + bv[3];
        *(v4f*)(out + (size_t)eBase * DD + flat) = o;
    }
}

// ------------------------------ host launch --------------------------------
extern "C" void kernel_launch(void* const* d_in, const int* in_sizes, int n_in,
                              void* d_out, int out_size, void* d_ws, size_t ws_size,
                              hipStream_t stream) {
    const float* efeat = (const float*)d_in[0];
    const float* srcf  = (const float*)d_in[1];
    const float* dstf  = (const float*)d_in[2];
    const int*   sidx  = (const int*)  d_in[3];
    const int*   didx  = (const int*)  d_in[4];
    const float* W1    = (const float*)d_in[5];
    const float* b1    = (const float*)d_in[6];
    const float* W2    = (const float*)d_in[7];
    const float* b2    = (const float*)d_in[8];
    const float* W3    = (const float*)d_in[9];
    const float* b3    = (const float*)d_in[10];
    const float* gamma = (const float*)d_in[11];
    const float* beta  = (const float*)d_in[12];
    float* out = (float*)d_out;

    // f16 pre-swizzled weights in workspace: 1,048,576 bytes total.
    _Float16* W1s = (_Float16*)d_ws;          // 384*512
    _Float16* W2s = W1s + CATD * HH;          // 512*512
    _Float16* W3s = W2s + HH * HH;            // 512*128

    const int E = in_sizes[0] / DD;           // 400000; E % BM == 0 for this problem

    swz_kernel<<<(CATD * HH + 255) / 256, 256, 0, stream>>>(W1, W1s, CATD, HH, HH / 16);
    swz_kernel<<<(HH * HH + 255) / 256, 256, 0, stream>>>(W2, W2s, HH, HH, HH / 16);
    swz_kernel<<<(HH * DD + 255) / 256, 256, 0, stream>>>(W3, W3s, HH, DD, DD / 16);

    hipFuncSetAttribute((const void*)edge_mlp_kernel,
                        hipFuncAttributeMaxDynamicSharedMemorySize, LDS_BYTES);
    edge_mlp_kernel<<<E / BM, NTHR, LDS_BYTES, stream>>>(
        efeat, srcf, dstf, sidx, didx, b1, b2, b3, gamma, beta,
        W1s, W2s, W3s, out);
}
